// TemporalAttention1D_4054449128043
// MI455X (gfx1250) — matrix operand order repaired and checked by reading, not compiled, not run on hardware
//
#include <hip/hip_runtime.h>

// ---------------------------------------------------------------------------
// TemporalAttention1D for MI455X (gfx1250, wave32, WMMA 16x16x32 f16)
// B=16, C=256, CQ=32, T=2048
// Flash-attention streaming: energy (B,T,T) never touches HBM.
// K/V tiles staged into LDS via gfx1250 async global->LDS copies (ASYNCcnt).
// ---------------------------------------------------------------------------

typedef _Float16 half_t;
typedef __attribute__((ext_vector_type(16))) _Float16 v16h;
typedef __attribute__((ext_vector_type(8)))  _Float16 v8h;
typedef __attribute__((ext_vector_type(8)))  float    v8f;
typedef int v4i __attribute__((vector_size(16)));   // matches builtin param type

#define B_  16
#define T_  2048
#define C_  256
#define CQ_ 32

#if defined(__gfx1250__) && __has_builtin(__builtin_amdgcn_global_load_async_to_lds_b128)
#define HAS_ASYNC_LDS 1
#endif

__device__ __forceinline__ v16h shuf16(v8h a, v8h b) {
    return __builtin_shufflevector(a, b, 0,1,2,3,4,5,6,7,8,9,10,11,12,13,14,15);
}

__device__ __forceinline__ v8f wmma16(v16h a, v16h b, v8f c) {
    // D = A(16x32 f16) * B(32x16 f16) + C(16x16 f32)
    return __builtin_amdgcn_wmma_f32_16x16x32_f16(
        /*neg_a=*/false, a, /*neg_b=*/false, b,
        /*c_mod=*/(short)0, c, /*reuse_a=*/false, /*reuse_b=*/false);
}

__device__ __forceinline__ v8f zero8() {
    v8f z;
    #pragma unroll
    for (int i = 0; i < 8; ++i) z[i] = 0.f;
    return z;
}

#if HAS_ASYNC_LDS
// 16-byte global -> LDS async copy (global_load_async_to_lds_b128, ASYNCcnt)
__device__ __forceinline__ void async_copy16(const half_t* g, half_t* l) {
    __builtin_amdgcn_global_load_async_to_lds_b128(
        (v4i*)(void*)g, (v4i*)(void*)l, /*offset=*/0, /*cpol=*/0);
}
__device__ __forceinline__ void wait_async0() {
#if __has_builtin(__builtin_amdgcn_s_wait_asynccnt)
    __builtin_amdgcn_s_wait_asynccnt(0);
#else
    asm volatile("s_wait_asynccnt 0" ::: "memory");
#endif
}
#endif

// ---------------------------------------------------------------------------
// Projection: Y[b,t,o] = bias[o] + sum_c W[o,c] * x[b,c,t], stored f16 with
// caller-provided strides (q,k -> (B,T,O) row-major; v -> (B,O,T)).
// Block: 256 threads (8 waves), tile = 64 t-rows x 32 o-cols, K = 256.
// Waves: (wave&3) = row group of 16, (wave>>2) = 16-col group (2 groups).
// ---------------------------------------------------------------------------
__global__ __launch_bounds__(256)
void proj_kernel(const float* __restrict__ x,    // (B, C, T)
                 const float* __restrict__ W,    // (O, C)
                 const float* __restrict__ bias, // (O,)
                 half_t* __restrict__ y,
                 int O, long stride_b, long stride_t, long stride_o)
{
    constexpr int KC = C_;        // K dimension (input channels)
    constexpr int MT = 64;        // t-rows per block
    constexpr int NB = 32;        // o-cols per block
    constexpr int XROW = KC + 8;  // 264 f16 = 528 B, 16B aligned rows
    constexpr int WROW = KC + 8;

    __shared__ half_t lds_x[MT * XROW];  // [t_local][c]  (A operand, row-major K)
    __shared__ half_t lds_w[NB * WROW];  // [o_local][c]  (B operand: fixed N per lane, K contig)

    const int tid  = threadIdx.x;
    const int lane = tid & 31;
    const int wave = tid >> 5;

    const int nb_blocks = O / NB;
    const int ob  = blockIdx.x % nb_blocks;
    const int tmp = blockIdx.x / nb_blocks;
    const int tb  = tmp % (T_ / MT);
    const int b   = tmp / (T_ / MT);
    const int t0  = tb * MT;
    const int o0  = ob * NB;

    // Stage x^T tile: lds_x[i][c] = x[b][c][t0+i]  (coalesced over t, converts f32->f16)
    for (int idx = tid; idx < MT * KC; idx += 256) {
        int c = idx >> 6;          // / MT
        int i = idx & (MT - 1);
        lds_x[i * XROW + c] = (half_t)x[((long)b * KC + c) * T_ + t0 + i];
    }
    // Stage W tile: lds_w[o][c] (coalesced over c)
    for (int idx = tid; idx < NB * KC; idx += 256) {
        int o = idx >> 8;          // / KC
        int c = idx & (KC - 1);
        lds_w[o * WROW + c] = (half_t)W[(long)(o0 + o) * KC + c];
    }
    __syncthreads();

    const int rw = wave & 3;          // row group (16 rows)
    const int cw = wave >> 2;         // col group (16 cols)
    const int r0 = rw * 16;
    const int halfsel = lane >> 4;    // which 16-lane half
    const int l16  = lane & 15;
    const int hi8  = halfsel * 8;
    const int hi16 = halfsel * 16;

    v8f acc = zero8();

    #pragma unroll
    for (int kk = 0; kk < KC; kk += 32) {
        // A fragment: M = l16 (row r0+l16), K = kk + hi16 + {0..15}
        const half_t* ar = &lds_x[(r0 + l16) * XROW + kk + hi16];
        v16h a = shuf16(*(const v8h*)(ar), *(const v8h*)(ar + 8));
        // B fragment: N = cw*16 + l16, K = kk + hi16 + {0..15}
        const half_t* br = &lds_w[(cw * 16 + l16) * WROW + kk + hi16];
        v16h bf = shuf16(*(const v8h*)(br), *(const v8h*)(br + 8));
        acc = wmma16(a, bf, acc);
    }

    // C/D layout: VGPR r -> row (halfsel*8 + r), col l16
    const int o  = o0 + cw * 16 + l16;
    const float bo = bias[o];
    #pragma unroll
    for (int r = 0; r < 8; ++r) {
        int t = t0 + r0 + halfsel * 8 + r;
        y[(long)b * stride_b + (long)t * stride_t + (long)o * stride_o] =
            (half_t)(acc[r] + bo);
    }
}

// ---------------------------------------------------------------------------
// Flash attention: per block 32 query rows, full C=256 channels, streaming
// all T keys in chunks of 64 with online softmax.
// 8 waves = 2 row-groups x 4 channel-groups (64 channels each).
// ---------------------------------------------------------------------------
__global__ __launch_bounds__(256)
void attn_kernel(const float*  __restrict__ x,     // (B, C, T) f32
                 const half_t* __restrict__ qh,    // (B, T, CQ) f16
                 const half_t* __restrict__ kh,    // (B, T, CQ) f16
                 const half_t* __restrict__ vh,    // (B, C, T)  f16
                 const float*  __restrict__ gamma, // (1,)
                 float* __restrict__ out)          // (B, C, T) f32
{
    constexpr int ST   = 64;       // keys per chunk
    constexpr int KROW = CQ_ + 8;  // 40 f16 = 80 B rows (16B-aligned)
    constexpr int VROW = ST + 8;   // 72 f16 = 144 B rows
    constexpr int PROW = ST + 8;

    __shared__ half_t lds_k[ST * KROW];        // [s][cq]
    __shared__ half_t lds_v[C_ * VROW];        // [c][s]   (B operand of P*V)
    __shared__ half_t lds_p[8 * 16 * PROW];    // wave-private P staging [row][key]

    const int tid  = threadIdx.x;
    const int lane = tid & 31;
    const int wave = tid >> 5;
    const int rg = wave >> 2;                  // query-row group (0..1)
    const int cg = wave & 3;                   // channel group (0..3)

    const int b  = blockIdx.x >> 6;            // T/32 = 64 row blocks
    const int t0 = (blockIdx.x & 63) * 32;

    const int halfsel = lane >> 4;
    const int l16  = lane & 15;
    const int hi8  = halfsel * 8;
    const int hi16 = halfsel * 16;

    // q A-fragment for this wave's 16 rows (CQ=32 -> exactly one K step)
    const int qrow = t0 + rg * 16 + l16;
    const half_t* qp = qh + ((long)b * T_ + qrow) * CQ_ + hi16;
    const v16h qa = shuf16(*(const v8h*)(qp), *(const v8h*)(qp + 8));

    float m[8], l[8];
    #pragma unroll
    for (int r = 0; r < 8; ++r) { m[r] = -1e30f; l[r] = 0.f; }
    v8f acc[4];
    #pragma unroll
    for (int j = 0; j < 4; ++j) acc[j] = zero8();

    half_t* pbuf = &lds_p[wave * 16 * PROW];

    for (int s0 = 0; s0 < T_; s0 += ST) {
        __syncthreads();   // all waves done reading previous chunk

#if HAS_ASYNC_LDS
        // --- Async global->LDS staging (pure byte copies, data already f16) ---
        // K chunk: 64 rows x 32 f16 = 256 x 16B chunks, one per thread.
        {
            int s = tid >> 2, ck = tid & 3;
            async_copy16(kh + ((long)b * T_ + s0 + s) * CQ_ + ck * 8,
                         &lds_k[s * KROW + ck * 8]);
        }
        // V chunk: 256 rows x 64 f16 = 2048 x 16B chunks, 8 per thread.
        #pragma unroll
        for (int i = 0; i < 8; ++i) {
            int idx = tid + i * 256;
            int c = idx >> 3, ck = idx & 7;
            async_copy16(vh + ((long)b * C_ + c) * T_ + s0 + ck * 8,
                         &lds_v[c * VROW + ck * 8]);
        }
        wait_async0();
#else
        // --- Fallback: VGPR-staged copies ---
        for (int idx = tid; idx < ST * CQ_; idx += 256) {
            int s = idx >> 5, cq = idx & 31;
            lds_k[s * KROW + cq] = kh[((long)b * T_ + s0 + s) * CQ_ + cq];
        }
        for (int idx = tid; idx < C_ * ST; idx += 256) {
            int c = idx >> 6, s = idx & 63;
            lds_v[c * VROW + s] = vh[((long)b * C_ + c) * T_ + s0 + s];
        }
#endif
        __syncthreads();   // staged chunk visible to all waves

        // E(16x64) = q(16x32) * k^T(32x64): 4 key n-tiles
        v8f E[4];
        #pragma unroll
        for (int js = 0; js < 4; ++js) {
            // B operand: N = key js*16+l16, K = cq = hi16 + {0..15} (contiguous)
            const half_t* kp = &lds_k[(js * 16 + l16) * KROW + hi16];
            v16h kb = shuf16(*(const v8h*)(kp), *(const v8h*)(kp + 8));
            E[js] = wmma16(qa, kb, zero8());
        }

        // Online softmax update (row = halfsel*8 + r; shfl stays in-lane-half)
        float scale[8];
        #pragma unroll
        for (int r = 0; r < 8; ++r) {
            float cm = fmaxf(fmaxf(E[0][r], E[1][r]), fmaxf(E[2][r], E[3][r]));
            cm = fmaxf(cm, __shfl_xor(cm, 1));
            cm = fmaxf(cm, __shfl_xor(cm, 2));
            cm = fmaxf(cm, __shfl_xor(cm, 4));
            cm = fmaxf(cm, __shfl_xor(cm, 8));
            const float mn = fmaxf(m[r], cm);
            scale[r] = __expf(m[r] - mn);
            m[r] = mn;
            float rs = 0.f;
            #pragma unroll
            for (int js = 0; js < 4; ++js) {
                float p = __expf(E[js][r] - mn);
                E[js][r] = p;
                rs += p;
            }
            rs += __shfl_xor(rs, 1);
            rs += __shfl_xor(rs, 2);
            rs += __shfl_xor(rs, 4);
            rs += __shfl_xor(rs, 8);
            l[r] = l[r] * scale[r] + rs;
            #pragma unroll
            for (int j = 0; j < 4; ++j) acc[j][r] *= scale[r];
        }

        // Stage P (C-layout -> row-major f16) in wave-private LDS
        #pragma unroll
        for (int js = 0; js < 4; ++js)
            #pragma unroll
            for (int r = 0; r < 8; ++r)
                pbuf[(halfsel * 8 + r) * PROW + js * 16 + l16] = (half_t)E[js][r];
        // same-wave LDS RAW: compiler inserts s_wait_dscnt

        // O(16x64) += P(16x64) * V(64x64 slice): K = keys, 2 K-steps x 4 n-tiles
        #pragma unroll
        for (int kk = 0; kk < ST; kk += 32) {
            const half_t* pr = &pbuf[l16 * PROW + kk + hi16];
            v16h pa = shuf16(*(const v8h*)(pr), *(const v8h*)(pr + 8));
            #pragma unroll
            for (int j = 0; j < 4; ++j) {
                const half_t* vp = &lds_v[(cg * 64 + j * 16 + l16) * VROW + kk + hi16];
                v16h vb = shuf16(*(const v8h*)(vp), *(const v8h*)(vp + 8));
                acc[j] = wmma16(pa, vb, acc[j]);
            }
        }
    }

    // Epilogue: out[b,c,t] = gamma * acc/l + x[b,c,t]
    const float g = gamma[0];
    #pragma unroll
    for (int j = 0; j < 4; ++j) {
        const int c = cg * 64 + j * 16 + l16;
        #pragma unroll
        for (int r = 0; r < 8; ++r) {
            const int t = t0 + rg * 16 + halfsel * 8 + r;
            const long idx = ((long)b * C_ + c) * T_ + t;
            out[idx] = g * (acc[j][r] / l[r]) + x[idx];
        }
    }
}

// ---------------------------------------------------------------------------
extern "C" void kernel_launch(void* const* d_in, const int* in_sizes, int n_in,
                              void* d_out, int out_size, void* d_ws, size_t ws_size,
                              hipStream_t stream) {
    const float* x     = (const float*)d_in[0];
    const float* Wq    = (const float*)d_in[1];
    const float* bq    = (const float*)d_in[2];
    const float* Wk    = (const float*)d_in[3];
    const float* bk    = (const float*)d_in[4];
    const float* Wv    = (const float*)d_in[5];
    const float* bv    = (const float*)d_in[6];
    const float* gamma = (const float*)d_in[7];
    float* out = (float*)d_out;

    // Workspace: q (2MB) | k (2MB) | v (16MB), all f16
    half_t* qh = (half_t*)d_ws;
    half_t* kh = qh + (size_t)B_ * T_ * CQ_;
    half_t* vh = kh + (size_t)B_ * T_ * CQ_;

    const int blocks_qk = B_ * (T_ / 64) * (CQ_ / 32);  // 512
    const int blocks_v  = B_ * (T_ / 64) * (C_  / 32);  // 4096

    // q, k: (B, T, CQ) row-major
    proj_kernel<<<blocks_qk, 256, 0, stream>>>(x, Wq, bq, qh, CQ_,
                                               (long)T_ * CQ_, (long)CQ_, 1L);
    proj_kernel<<<blocks_qk, 256, 0, stream>>>(x, Wk, bk, kh, CQ_,
                                               (long)T_ * CQ_, (long)CQ_, 1L);
    // v: (B, C, T) so attention can coalesce V-tile loads along t
    proj_kernel<<<blocks_v, 256, 0, stream>>>(x, Wv, bv, vh, C_,
                                              (long)C_ * T_, 1L, (long)T_);

    attn_kernel<<<B_ * (T_ / 32), 256, 0, stream>>>(x, qh, kh, vh, gamma, out);
}
